// GCNSyntaxGuidedTransformerEncoderLayer_56684978373134
// MI455X (gfx1250) — matrix operand order repaired and checked by reading, not compiled, not run on hardware
//
#include <hip/hip_runtime.h>
#include <cstdint>
#include <cstddef>

// ---------------------------------------------------------------------------
// GCN syntax-guided transformer encoder layer for MI455X (gfx1250, wave32).
// Heavy einsums run as bf16 WMMA (v_wmma_f32_16x16x32_bf16, f32 accumulate).
// The (B,S,S,D) ReLU intermediates are never materialized: the P-weighted
// s-reduction is fused into the GEMM epilogue via LDS float atomics.
// ---------------------------------------------------------------------------

typedef unsigned short u16;
typedef unsigned int   u32;

typedef __attribute__((ext_vector_type(16))) __bf16 v16bf;
typedef __attribute__((ext_vector_type(8)))  float  v8f;

#define DD  512
#define SS  128
#define BB  4
#define FFN 2048

__device__ __forceinline__ u16 f2bf(float x) {
  u32 u = __float_as_uint(x);
  u32 r = (u + 0x7FFFu + ((u >> 16) & 1u)) >> 16;   // round-to-nearest-even
  return (u16)r;
}
__device__ __forceinline__ u32 pack2(float a, float b) {
  return (u32)f2bf(a) | ((u32)f2bf(b) << 16);
}

union Frag {
  v16bf bf;
  uint4 q[2];
};

__device__ __forceinline__ v8f v8f_zero() {
  v8f z;
#pragma unroll
  for (int i = 0; i < 8; ++i) z[i] = 0.0f;
  return z;
}

constexpr int KC   = 32;   // K chunk per LDS stage (one wmma K depth)
constexpr int LDA_ = 40;   // padded LDS row stride in u16 (80B, conflict-light)

// ---------------------------------------------------------------------------
// Fused arc-GEMM + ReLU + P-weighted aggregation.
// Grid: B*S blocks (one per (b,t)); 256 threads = 8 waves.
// pass 0: U_in  = relu(h_in[b,s,e]  + arc_in[b,t,s,:]@Wi_a^T  + b_in ), w = P[b,t,s]
// pass 1: U_out = relu(h_out[b,s,e] + arc_out[b,t,s,:]@Wo_a^T + b_out), w = P[b,s,t]
// agg[b,t,e] = sum_s w*U   (both passes accumulated in LDS)
// ---------------------------------------------------------------------------
__global__ __launch_bounds__(256) void arc_agg_kernel(
    const float* __restrict__ arc_in, const float* __restrict__ arc_out,
    const float* __restrict__ P,
    const u16* __restrict__ wi_a, const u16* __restrict__ wo_a,
    const float* __restrict__ h_in, const float* __restrict__ h_out,
    const float* __restrict__ b_in, const float* __restrict__ b_out,
    float* __restrict__ agg)
{
  __shared__ u16   lds_a[128 * LDA_];   // arc slab, bf16 (s x k)
  __shared__ u16   lds_b[256 * LDA_];   // weight slab, bf16 (e x k)
  __shared__ float agg_lds[DD];

  const int bt   = blockIdx.x;
  const int b    = bt >> 7;           // / S
  const int t    = bt & (SS - 1);
  const int tid  = threadIdx.x;
  const int wave = tid >> 5;
  const int lane = tid & 31;
  const int half = lane >> 4;
  const int l16  = lane & 15;

  for (int i = tid; i < DD; i += 256) agg_lds[i] = 0.0f;
  __syncthreads();

  for (int pass = 0; pass < 2; ++pass) {
    const float* __restrict__ arcBT =
        (pass ? arc_out : arc_in) + (size_t)(b * SS + t) * SS * DD;
    const u16*   __restrict__ W     = pass ? wo_a  : wi_a;
    const float* __restrict__ hterm = pass ? h_out : h_in;
    const float* __restrict__ bias  = pass ? b_out : b_in;

    // P weights for this wave's 16 s-rows (C layout: M = v + 8*half)
    float pv[8];
#pragma unroll
    for (int v = 0; v < 8; ++v) {
      int s = wave * 16 + half * 8 + v;
      pv[v] = pass ? P[(size_t)(b * SS + s) * SS + t]     // P[b,s,t]
                   : P[(size_t)(b * SS + t) * SS + s];    // P[b,t,s]
    }

    for (int nh = 0; nh < 2; ++nh) {   // N (=e) processed in halves of 256
      v8f acc[16];
#pragma unroll
      for (int ct = 0; ct < 16; ++ct) acc[ct] = v8f_zero();

      for (int kk = 0; kk < DD; kk += KC) {
        // ---- stage A: 128x32 fp32 arc -> bf16 LDS (1024 float4 loads) ----
#pragma unroll
        for (int it = 0; it < 4; ++it) {
          int idx = tid + it * 256;          // 0..1023
          int row = idx >> 3;
          int kc4 = (idx & 7) << 2;
          const float4 f = *(const float4*)(arcBT + (size_t)row * DD + kk + kc4);
          uint2 pk;
          pk.x = pack2(f.x, f.y);
          pk.y = pack2(f.z, f.w);
          *(uint2*)&lds_a[row * LDA_ + kc4] = pk;
          if (kk + KC < DD)                  // prefetch next K slab
            __builtin_prefetch(arcBT + (size_t)row * DD + kk + KC + kc4, 0, 0);
        }
        // ---- stage B: 256x32 bf16 weights (L2-resident) ----
#pragma unroll
        for (int it = 0; it < 4; ++it) {
          int idx = tid + it * 256;
          int row = idx >> 2;
          int kc8 = (idx & 3) << 3;
          uint4 q = *(const uint4*)(W + (size_t)(nh * 256 + row) * DD + kk + kc8);
          *(uint4*)&lds_b[row * LDA_ + kc8] = q;
        }
        __syncthreads();

        // A fragment: lane = M row (l16); K = {kb..kb+7, kb+16..kb+23}, kb=8*half
        Frag a;
        const u16* ap = &lds_a[(wave * 16 + l16) * LDA_ + half * 8];
        a.q[0] = *(const uint4*)ap;
        a.q[1] = *(const uint4*)(ap + 16);
#pragma unroll
        for (int ct = 0; ct < 16; ++ct) {
          // B fragment: lane = N col (l16); K = {kb..kb+7, kb+16..kb+23}, kb=8*half
          Frag bb_;
          const u16* bp = &lds_b[(ct * 16 + l16) * LDA_ + half * 8];
          bb_.q[0] = *(const uint4*)bp;
          bb_.q[1] = *(const uint4*)(bp + 16);
          acc[ct] = __builtin_amdgcn_wmma_f32_16x16x32_bf16(
              false, a.bf, false, bb_.bf, (short)0, acc[ct], false, false);
        }
        __syncthreads();
      }

      // ---- epilogue: +h_term +bias, ReLU, P-weighted s-reduction ----
#pragma unroll
      for (int ct = 0; ct < 16; ++ct) {
        int   e  = nh * 256 + ct * 16 + l16;
        float bs = bias[e];
        float part = 0.0f;
#pragma unroll
        for (int v = 0; v < 8; ++v) {
          int s = wave * 16 + half * 8 + v;
          float u = acc[ct][v] + hterm[(size_t)(b * SS + s) * DD + e] + bs;
          u = fmaxf(u, 0.0f);
          part += pv[v] * u;
        }
        atomicAdd(&agg_lds[e], part);      // ds_add_f32
      }
      __syncthreads();
    }
  }

  for (int i = tid; i < DD; i += 256)
    agg[(size_t)(b * SS + t) * DD + i] = agg_lds[i];
}

// ---------------------------------------------------------------------------
// Generic bf16 WMMA GEMM: C[m,n] = sum_k A[m,k]*Bw[n,k]  (both K-contiguous)
// Block computes a 128x128 tile; 8 waves, each 1 row-tile x 8 col-tiles.
// FLAGS: 1=+bias[n]  2=ReLU  4=+resid[m,n]  8=write bf16  16=write f32
// ---------------------------------------------------------------------------
template <int FLAGS>
__global__ __launch_bounds__(256) void gemm_bf16_kernel(
    const u16* __restrict__ A, const u16* __restrict__ Bw,
    const float* __restrict__ bias, const float* __restrict__ resid,
    float* __restrict__ Cf, u16* __restrict__ Cbf, int M, int N, int K)
{
  __shared__ u16 lds_a[128 * LDA_];
  __shared__ u16 lds_b[128 * LDA_];

  const int n0   = blockIdx.x * 128;
  const int m0   = blockIdx.y * 128;
  const int tid  = threadIdx.x;
  const int wave = tid >> 5;
  const int lane = tid & 31;
  const int half = lane >> 4;
  const int l16  = lane & 15;

  v8f acc[8];
#pragma unroll
  for (int ct = 0; ct < 8; ++ct) acc[ct] = v8f_zero();

  for (int kk = 0; kk < K; kk += KC) {
#pragma unroll
    for (int it = 0; it < 2; ++it) {
      int idx = tid + it * 256;   // 0..511
      int row = idx >> 2;
      int kc8 = (idx & 3) << 3;
      *(uint4*)&lds_a[row * LDA_ + kc8] =
          *(const uint4*)(A + (size_t)(m0 + row) * K + kk + kc8);
      *(uint4*)&lds_b[row * LDA_ + kc8] =
          *(const uint4*)(Bw + (size_t)(n0 + row) * K + kk + kc8);
    }
    __syncthreads();

    Frag a;
    const u16* ap = &lds_a[(wave * 16 + l16) * LDA_ + half * 8];
    a.q[0] = *(const uint4*)ap;
    a.q[1] = *(const uint4*)(ap + 16);
#pragma unroll
    for (int ct = 0; ct < 8; ++ct) {
      Frag bb_;
      const u16* bp = &lds_b[(ct * 16 + l16) * LDA_ + half * 8];
      bb_.q[0] = *(const uint4*)bp;
      bb_.q[1] = *(const uint4*)(bp + 16);
      acc[ct] = __builtin_amdgcn_wmma_f32_16x16x32_bf16(
          false, a.bf, false, bb_.bf, (short)0, acc[ct], false, false);
    }
    __syncthreads();
  }

#pragma unroll
  for (int ct = 0; ct < 8; ++ct) {
    int   n  = n0 + ct * 16 + l16;
    float bs = (FLAGS & 1) ? bias[n] : 0.0f;
#pragma unroll
    for (int v = 0; v < 8; ++v) {
      int m = m0 + wave * 16 + half * 8 + v;
      float val = acc[ct][v] + bs;
      if (FLAGS & 4)  val += resid[(size_t)m * N + n];
      if (FLAGS & 2)  val = fmaxf(val, 0.0f);
      if (FLAGS & 16) Cf[(size_t)m * N + n] = val;
      if (FLAGS & 8)  Cbf[(size_t)m * N + n] = f2bf(val);
    }
  }
}

// ---------------------------------------------------------------------------
// LayerNorm over D=512. Row r = s*B + b of an (S,B,D) tensor.
// Optionally adds transposed agg[(b*S+s)*D + e] before normalizing.
// Writes f32 and (optionally) bf16 copies.
// ---------------------------------------------------------------------------
__global__ __launch_bounds__(256) void layernorm_kernel(
    const float* __restrict__ X, const float* __restrict__ Agg,
    const float* __restrict__ w, const float* __restrict__ bv,
    float* __restrict__ Yf, u16* __restrict__ Ybf)
{
  __shared__ float red[256], red2[256];
  const int r   = blockIdx.x;     // s*B + b
  const int s   = r >> 2;         // / B  (B = 4)
  const int b   = r & 3;
  const int tid = threadIdx.x;

  const size_t base  = (size_t)r * DD;
  const size_t abase = (size_t)(b * SS + s) * DD;

  float x0 = X[base + tid];
  float x1 = X[base + tid + 256];
  if (Agg) { x0 += Agg[abase + tid]; x1 += Agg[abase + tid + 256]; }

  red[tid]  = x0 + x1;
  red2[tid] = x0 * x0 + x1 * x1;
  __syncthreads();
  for (int st = 128; st > 0; st >>= 1) {
    if (tid < st) { red[tid] += red[tid + st]; red2[tid] += red2[tid + st]; }
    __syncthreads();
  }
  const float mu  = red[0] * (1.0f / DD);
  const float var = red2[0] * (1.0f / DD) - mu * mu;
  const float rs  = rsqrtf(var + 1e-5f);

  float y0 = (x0 - mu) * rs * w[tid]       + bv[tid];
  float y1 = (x1 - mu) * rs * w[tid + 256] + bv[tid + 256];
  Yf[base + tid]       = y0;
  Yf[base + tid + 256] = y1;
  if (Ybf) { Ybf[base + tid] = f2bf(y0); Ybf[base + tid + 256] = f2bf(y1); }
}

// ------------------------- prep / conversion kernels -----------------------

// W (D x 2D) -> Wh = bf16(W[:, :D]), Wa = bf16(W[:, D:])  (both D x D, k-major)
__global__ __launch_bounds__(256) void split_w_kernel(
    const float* __restrict__ W, u16* __restrict__ Wh, u16* __restrict__ Wa)
{
  int i = blockIdx.x * 256 + threadIdx.x;       // over D*D
  int e = i >> 9, d = i & (DD - 1);
  Wh[i] = f2bf(W[(size_t)e * (2 * DD) + d]);
  Wa[i] = f2bf(W[(size_t)e * (2 * DD) + DD + d]);
}

__global__ __launch_bounds__(256) void conv_bf_kernel(
    const float* __restrict__ src, u16* __restrict__ dst)
{
  int i = blockIdx.x * 256 + threadIdx.x;
  dst[i] = f2bf(src[i]);
}

// hb_bf[(b*S+s)*D + d] = bf16(h[(s*B+b)*D + d])   (transpose S,B -> B,S)
__global__ __launch_bounds__(256) void hb_conv_kernel(
    const float* __restrict__ h, u16* __restrict__ hb)
{
  int i  = blockIdx.x * 256 + threadIdx.x;      // over B*S*D
  int d  = i & (DD - 1);
  int bs = i >> 9;
  int b  = bs >> 7;
  int s  = bs & (SS - 1);
  hb[i] = f2bf(h[(size_t)(s * BB + b) * DD + d]);
}

// ------------------------------ workspace map ------------------------------
constexpr size_t OFF_WO_H = 0;                          // 512*512*2
constexpr size_t OFF_WO_A = OFF_WO_H + 524288;
constexpr size_t OFF_WI_H = OFF_WO_A + 524288;
constexpr size_t OFF_WI_A = OFF_WI_H + 524288;
constexpr size_t OFF_FC1  = OFF_WI_A + 524288;          // 2048*512*2
constexpr size_t OFF_FC2  = OFF_FC1  + 2097152;         // 512*2048*2
constexpr size_t OFF_HB   = OFF_FC2  + 2097152;         // 4*128*512*2
constexpr size_t OFF_HOUT = OFF_HB   + 524288;          // 512*512*4
constexpr size_t OFF_HIN  = OFF_HOUT + 1048576;
constexpr size_t OFF_AGG  = OFF_HIN  + 1048576;
constexpr size_t OFF_X1F  = OFF_AGG  + 1048576;
constexpr size_t OFF_X1BF = OFF_X1F  + 1048576;         // 512*512*2
constexpr size_t OFF_YBF  = OFF_X1BF + 524288;          // 512*2048*2
constexpr size_t OFF_PRE2 = OFF_YBF  + 2097152;         // 512*512*4

extern "C" void kernel_launch(void* const* d_in, const int* in_sizes, int n_in,
                              void* d_out, int out_size, void* d_ws, size_t ws_size,
                              hipStream_t stream) {
  (void)in_sizes; (void)n_in; (void)out_size; (void)ws_size;

  const float* h       = (const float*)d_in[0];   // (S,B,D)
  const float* arc_out = (const float*)d_in[1];   // (B,S,S,D)
  const float* arc_in  = (const float*)d_in[2];   // (B,S,S,D)
  const float* P       = (const float*)d_in[3];   // (B,S,S)
  const float* W_out_w = (const float*)d_in[4];
  const float* W_out_b = (const float*)d_in[5];
  const float* W_in_w  = (const float*)d_in[6];
  const float* W_in_b  = (const float*)d_in[7];
  const float* fc1_w   = (const float*)d_in[8];
  const float* fc1_b   = (const float*)d_in[9];
  const float* fc2_w   = (const float*)d_in[10];
  const float* fc2_b   = (const float*)d_in[11];
  const float* ln1_w   = (const float*)d_in[12];
  const float* ln1_b   = (const float*)d_in[13];
  const float* ln2_w   = (const float*)d_in[14];
  const float* ln2_b   = (const float*)d_in[15];
  float* out = (float*)d_out;

  char* ws = (char*)d_ws;
  u16*   wo_h   = (u16*)(ws + OFF_WO_H);
  u16*   wo_a   = (u16*)(ws + OFF_WO_A);
  u16*   wi_h   = (u16*)(ws + OFF_WI_H);
  u16*   wi_a   = (u16*)(ws + OFF_WI_A);
  u16*   fc1_bf = (u16*)(ws + OFF_FC1);
  u16*   fc2_bf = (u16*)(ws + OFF_FC2);
  u16*   hb_bf  = (u16*)(ws + OFF_HB);
  float* h_outf = (float*)(ws + OFF_HOUT);
  float* h_inf  = (float*)(ws + OFF_HIN);
  float* agg_f  = (float*)(ws + OFF_AGG);
  float* x1_f   = (float*)(ws + OFF_X1F);
  u16*   x1_bf  = (u16*)(ws + OFF_X1BF);
  u16*   y_bf   = (u16*)(ws + OFF_YBF);
  float* pre2_f = (float*)(ws + OFF_PRE2);

  // --- prep: bf16 conversions (weights stay L2-resident afterwards) ---
  split_w_kernel<<<(DD * DD) / 256, 256, 0, stream>>>(W_out_w, wo_h, wo_a);
  split_w_kernel<<<(DD * DD) / 256, 256, 0, stream>>>(W_in_w,  wi_h, wi_a);
  conv_bf_kernel<<<(FFN * DD) / 256, 256, 0, stream>>>(fc1_w, fc1_bf);
  conv_bf_kernel<<<(DD * FFN) / 256, 256, 0, stream>>>(fc2_w, fc2_bf);
  hb_conv_kernel<<<(BB * SS * DD) / 256, 256, 0, stream>>>(h, hb_bf);

  // --- h projections: h_out/h_in = hb @ W_{o,i}_h^T  (M=N=K=512) ---
  dim3 gP(DD / 128, (BB * SS) / 128);
  gemm_bf16_kernel<16><<<gP, 256, 0, stream>>>(hb_bf, wo_h, nullptr, nullptr,
                                               h_outf, nullptr, BB * SS, DD, DD);
  gemm_bf16_kernel<16><<<gP, 256, 0, stream>>>(hb_bf, wi_h, nullptr, nullptr,
                                               h_inf, nullptr, BB * SS, DD, DD);

  // --- fused arc GEMM + ReLU + P-weighted aggregation (the 68.7 GFLOP) ---
  arc_agg_kernel<<<BB * SS, 256, 0, stream>>>(arc_in, arc_out, P, wi_a, wo_a,
                                              h_inf, h_outf, W_in_b, W_out_b,
                                              agg_f);

  // --- LN1 on residual + agg^T ---
  layernorm_kernel<<<SS * BB, 256, 0, stream>>>(h, agg_f, ln1_w, ln1_b,
                                                x1_f, x1_bf);

  // --- FFN: y = relu(x1 @ fc1^T + b1); pre2 = y @ fc2^T + b2 + x1 ---
  dim3 g1(FFN / 128, (SS * BB) / 128);
  gemm_bf16_kernel<1 | 2 | 8><<<g1, 256, 0, stream>>>(
      x1_bf, fc1_bf, fc1_b, nullptr, nullptr, y_bf, SS * BB, FFN, DD);
  dim3 g2(DD / 128, (SS * BB) / 128);
  gemm_bf16_kernel<1 | 4 | 16><<<g2, 256, 0, stream>>>(
      y_bf, fc2_bf, fc2_b, x1_f, pre2_f, nullptr, SS * BB, DD, FFN);

  // --- LN2 -> output (S,B,D) f32 ---
  layernorm_kernel<<<SS * BB, 256, 0, stream>>>(pre2_f, nullptr, ln2_w, ln2_b,
                                                out, nullptr);
}
